// TransformerBlock_66941360275876
// MI455X (gfx1250) — matrix operand order repaired and checked by reading, not compiled, not run on hardware
//
#include <hip/hip_runtime.h>
#include <hip/hip_bf16.h>
#include <cmath>

typedef __attribute__((ext_vector_type(16))) _Float16 v16h;
typedef __attribute__((ext_vector_type(8)))  float    v8f;
typedef __attribute__((ext_vector_type(4)))  float    v4f;
typedef unsigned int v4u __attribute__((ext_vector_type(4)));
typedef int          v8i __attribute__((ext_vector_type(8)));
typedef int          v4i __attribute__((ext_vector_type(4)));

#define Bn 2
#define Dc 64
#define NHEADS 2
#define HID 128
#define MSEL 32
#define HW 4096
#define PC 192
#define PH 512
#define PW 512

#if defined(__has_builtin)
#if __has_builtin(__builtin_amdgcn_tensor_load_to_lds) && __has_builtin(__builtin_amdgcn_s_wait_tensorcnt)
#define USE_TDM 1
#endif
#endif
#ifndef USE_TDM
#define USE_TDM 0
#endif

// ======================= generic WMMA GEMM =======================
// C[b] = A[b](MxK) * B[b](KxN, or NxK if BT) [* Bmul elementwise] + bias
// One wave per 16x16 C tile. BT/MUL are compile-time so the k-loop is pure
// b128 loads + cvt + v_wmma_f32_16x16x32_f16.
template <int BT, int MUL>
__global__ __launch_bounds__(32)
void gemm_wmma_kernel(const float* __restrict__ A, const float* __restrict__ B,
                      const float* __restrict__ Bmul, float* __restrict__ C,
                      const float* __restrict__ bias,
                      int Ndim, int Kdim, long sA, long sB, long sC)
{
    const int tn = blockIdx.x, tm = blockIdx.y, bz = blockIdx.z;
    const float* Ab  = A + (long)bz * sA;
    const float* Bb  = B + (long)bz * sB;
    const float* Bmb = Bmul + (long)bz * sB;   // only dereferenced if MUL
    float*       Cb  = C + (long)bz * sC;
    const int lane = threadIdx.x & 31;
    const int nloc = lane & 15;
    const int half = lane >> 4;
    const int arow = tm * 16 + nloc;
    const int ncol = tn * 16 + nloc;
    v8f acc = {};
    for (int k0 = 0; k0 < Kdim; k0 += 32) {
        v16h afr, bfr;
        // A fragment: rows of K, one contiguous 16-float run per lane
        {
            const float* ap = Ab + (long)arow * Kdim + k0 + 16 * half;
            v4f a0 = *(const v4f*)(ap);
            v4f a1 = *(const v4f*)(ap + 4);
            v4f a2 = *(const v4f*)(ap + 8);
            v4f a3 = *(const v4f*)(ap + 12);
#pragma unroll
            for (int j = 0; j < 4; ++j) {
                afr[j]      = (_Float16)a0[j];
                afr[4 + j]  = (_Float16)a1[j];
                afr[8 + j]  = (_Float16)a2[j];
                afr[12 + j] = (_Float16)a3[j];
            }
        }
        if constexpr (BT) {
            // B stored N x K: 16 contiguous floats per lane
            const float* bp = Bb + (long)ncol * Kdim + k0 + 16 * half;
            v4f b0 = *(const v4f*)(bp);
            v4f b1 = *(const v4f*)(bp + 4);
            v4f b2 = *(const v4f*)(bp + 8);
            v4f b3 = *(const v4f*)(bp + 12);
            if constexpr (MUL) {
                const float* mp = Bmb + (long)ncol * Kdim + k0 + 16 * half;
                b0 *= *(const v4f*)(mp);
                b1 *= *(const v4f*)(mp + 4);
                b2 *= *(const v4f*)(mp + 8);
                b3 *= *(const v4f*)(mp + 12);
            }
#pragma unroll
            for (int j = 0; j < 4; ++j) {
                bfr[j]      = (_Float16)b0[j];
                bfr[4 + j]  = (_Float16)b1[j];
                bfr[8 + j]  = (_Float16)b2[j];
                bfr[12 + j] = (_Float16)b3[j];
            }
        } else {
            // B stored K x N: one column per lane (coalesced across lanes)
            const float* bp = Bb + (long)(k0 + 16 * half) * Ndim + ncol;
#pragma unroll
            for (int j = 0; j < 16; ++j) {
                float v = bp[(long)j * Ndim];
                if constexpr (MUL) v *= Bmb[(long)(k0 + 16 * half + j) * Ndim + ncol];
                bfr[j] = (_Float16)v;
            }
        }
        acc = __builtin_amdgcn_wmma_f32_16x16x32_f16(false, afr, false, bfr,
                                                     (short)0, acc, false, false);
    }
#pragma unroll
    for (int r = 0; r < 8; ++r) {
        int m = tm * 16 + r + 8 * half;
        float v = acc[r];
        if (bias) v += bias[m];
        Cb[(long)m * Ndim + ncol] = v;
    }
}

// ======== fused perception 1x1 conv (both weight sets) + 8x8 max/avg pool ========
// TDM (tensor_load_to_lds) DMAs the 8x8x192 f32 window tile into LDS while the
// VALU stages the f16 weight stack; WMMA consumes both, then in-register
// max/sum pooling produces p1/p2 for both matching blocks in one pass.
__global__ __launch_bounds__(128)
void perc_fused_kernel(const float* __restrict__ perc,
                       const float* __restrict__ Wa, const float* __restrict__ biasA,
                       const float* __restrict__ Wf, const float* __restrict__ biasF,
                       float* __restrict__ p1a, float* __restrict__ p2a,
                       float* __restrict__ p1f, float* __restrict__ p2f)
{
    __shared__ _Float16 Ws[128][PC];   // stacked weights (attn rows 0..63, ffn 64..127)
    __shared__ float    Xs[PC][64];    // 192 channels x 64 window positions (f32)
    __shared__ float sMax[4][128];
    __shared__ float sSum[4][128];

    const int tid  = threadIdx.x;
    const int wave = tid >> 5;
    const int lane = tid & 31;
    const int win  = blockIdx.x;            // b*4096 + wy*64 + wx
    const int b  = win >> 12;
    const int wy = (win >> 6) & 63;
    const int wx = win & 63;
    const float* pb = perc + (long)b * PC * PH * PW;

#if USE_TDM
    if (wave == 0) {
        // Tensor DMA descriptor (D#): 3D tile 8(x) x 8(y) x 192(c) of f32,
        // tensor strides 512 / 512*512, contiguous into LDS as Xs[c][y*8+x].
        unsigned lds_addr = (unsigned)(unsigned long long)
            (__attribute__((address_space(3))) const void*)&Xs[0][0];
        unsigned long long ga = (unsigned long long)(const void*)
            (pb + ((long)(wy * 8)) * PW + wx * 8);
        v4u g0;
        g0[0] = 1u;                                   // count=1 (valid), user mode
        g0[1] = lds_addr;                             // LDS byte address
        g0[2] = (unsigned)(ga & 0xFFFFFFFFu);         // global_addr[31:0]
        g0[3] = (unsigned)((ga >> 32) & 0x01FFFFFFu)  // global_addr[56:32]
                | (2u << 30);                         // type = 2 ("image")
        v8i g1;
        g1[0] = (2 << 16);            // data_size = 4B
        g1[1] = (PW << 16);           // tensor_dim0[15:0] -> bits[31:16]
        g1[2] = (PH << 16);           // tensor_dim1[15:0] -> bits[31:16]
        g1[3] = (8 << 16);            // tile_dim0 = 8
        g1[4] = (PC << 16) | 8;       // tile_dim1 = 8, tile_dim2 = 192
        g1[5] = PW;                   // tensor_dim0_stride[31:0] = 512
        g1[6] = 0;                    // stride0[47:32]=0, stride1[15:0]=0
        g1[7] = (PH * PW) >> 16;      // tensor_dim1_stride[47:16] (512*512)
        v4i g2;
        g2[0] = PC;                   // tensor_dim2 = 192
        g2[1] = 0;                    // tensor_dim3
        g2[2] = PH * PW;              // tensor_dim2_stride[31:0]
        g2[3] = 0;                    // stride2[47:32]=0, tile_dim3=0
        v4i g3 = {0, 0, 0, 0};
        v8i g4 = {0, 0, 0, 0, 0, 0, 0, 0};
        __builtin_amdgcn_tensor_load_to_lds(g0, g1, g2, g3, g4, 0);
    }
#endif

    // weights: 128 rows x 48 float4 quads (overlaps with TDM in flight)
    for (int e = tid; e < 128 * (PC / 4); e += 128) {
        int r = e / (PC / 4), kq = e % (PC / 4);
        const float* src = (r < 64) ? (Wa + (long)r * PC) : (Wf + (long)(r - 64) * PC);
        v4f wv = *(const v4f*)(src + kq * 4);
#pragma unroll
        for (int j = 0; j < 4; ++j) Ws[r][kq * 4 + j] = (_Float16)wv[j];
    }
#if USE_TDM
    if (wave == 0) __builtin_amdgcn_s_wait_tensorcnt(0);
#else
    // fallback staging: 192 channels x 8 rows x 2 float4 quads
    for (int e = tid; e < PC * 16; e += 128) {
        int c = e >> 4, rem = e & 15;
        int dy = rem >> 1, dxq = rem & 1;
        v4f v = *(const v4f*)(pb + ((long)c * PH + (wy * 8 + dy)) * PW + wx * 8 + dxq * 4);
#pragma unroll
        for (int j = 0; j < 4; ++j) Xs[c][dy * 8 + dxq * 4 + j] = v[j];
    }
#endif
    __syncthreads();

    const int nloc = lane & 15;
    const int half = lane >> 4;
    v8f acc[8] = {};
    for (int ks = 0; ks < PC; ks += 32) {
        v16h bfr;
#pragma unroll
        for (int j = 0; j < 16; ++j)
            bfr[j] = (_Float16)Xs[ks + 16 * half + j][wave * 16 + nloc];
#pragma unroll
        for (int mt = 0; mt < 8; ++mt) {
            v16h afr;
#pragma unroll
            for (int j = 0; j < 16; ++j) {
                int kk = ks + 16 * half + j;
                afr[j] = Ws[mt * 16 + nloc][kk];
            }
            acc[mt] = __builtin_amdgcn_wmma_f32_16x16x32_f16(false, afr, false, bfr,
                                                             (short)0, acc[mt], false, false);
        }
    }
    // reduce over this wave's 16 window positions (lane bits 0..3)
#pragma unroll
    for (int mt = 0; mt < 8; ++mt) {
#pragma unroll
        for (int r = 0; r < 8; ++r) {
            float mx = acc[mt][r];
            float sm = acc[mt][r];
#pragma unroll
            for (int o = 1; o < 16; o <<= 1) {
                mx = fmaxf(mx, __shfl_xor(mx, o, 32));
                sm += __shfl_xor(sm, o, 32);
            }
            if (nloc == 0) {
                int row = mt * 16 + 8 * half + r;
                sMax[wave][row] = mx;
                sSum[wave][row] = sm;
            }
        }
    }
    __syncthreads();
    {
        int row = tid;  // 0..127
        float mx = fmaxf(fmaxf(sMax[0][row], sMax[1][row]), fmaxf(sMax[2][row], sMax[3][row]));
        float sm = sSum[0][row] + sSum[1][row] + sSum[2][row] + sSum[3][row];
        int owin = wy * 64 + wx;
        if (row < 64) {
            p1a[((long)b * 64 + row) * HW + owin] = mx + biasA[row];
            p2a[((long)b * 64 + row) * HW + owin] = sm * (1.0f / 64.0f) + biasA[row];
        } else {
            int r2 = row - 64;
            p1f[((long)b * 64 + r2) * HW + owin] = mx + biasF[r2];
            p2f[((long)b * 64 + r2) * HW + owin] = sm * (1.0f / 64.0f) + biasF[r2];
        }
    }
}

// ======================= small helper kernels =======================
__global__ __launch_bounds__(256)
void layernorm_kernel(const float* __restrict__ x, const float* __restrict__ w,
                      const float* __restrict__ bb, float* __restrict__ y)
{
    int p = blockIdx.x * blockDim.x + threadIdx.x;   // B*HW positions
    if (p >= Bn * HW) return;
    int b = p / HW, hw = p % HW;
    const float* xp = x + (long)b * Dc * HW + hw;
    float mu = 0.f;
    for (int c = 0; c < Dc; ++c) mu += xp[(long)c * HW];
    mu *= (1.0f / Dc);
    float var = 0.f;
    for (int c = 0; c < Dc; ++c) { float d = xp[(long)c * HW] - mu; var += d * d; }
    var *= (1.0f / Dc);
    float inv = rsqrtf(var + 1e-5f);
    float* yp = y + (long)b * Dc * HW + hw;
    for (int c = 0; c < Dc; ++c) yp[(long)c * HW] = (xp[(long)c * HW] - mu) * inv * w[c] + bb[c];
}

// grouped 3x3 conv, pad=1, optional exact GELU
__global__ __launch_bounds__(256)
void gconv3_kernel(const float* __restrict__ x, const float* __restrict__ w,
                   const float* __restrict__ bias, float* __restrict__ y,
                   int Cin, int Cout, int G, int act)
{
    long n = (long)blockIdx.x * blockDim.x + threadIdx.x;
    long tot = (long)Bn * Cout * HW;
    if (n >= tot) return;
    int hw = (int)(n % HW);
    int co = (int)((n / HW) % Cout);
    int b  = (int)(n / ((long)Cout * HW));
    int yy = hw >> 6, xx = hw & 63;
    int cog = Cout / G, cig = Cin / G;
    int g = co / cog;
    float acc = bias[co];
    for (int ci = 0; ci < cig; ++ci) {
        int cin = g * cig + ci;
        const float* xp = x + ((long)b * Cin + cin) * HW;
        const float* wp = w + ((long)co * cig + ci) * 9;
        for (int ky = 0; ky < 3; ++ky) {
            int iy = yy + ky - 1; if (iy < 0 || iy > 63) continue;
            for (int kx = 0; kx < 3; ++kx) {
                int ix = xx + kx - 1; if (ix < 0 || ix > 63) continue;
                acc += wp[ky * 3 + kx] * xp[iy * 64 + ix];
            }
        }
    }
    if (act) acc = 0.5f * acc * (1.0f + erff(acc * 0.70710678f));
    y[n] = acc;
}

// squared L2 norm of each HW-length row; rows = B*C, per-batch base stride
__global__ __launch_bounds__(256)
void rownorm2_kernel(const float* __restrict__ X, float* __restrict__ out,
                     int C, long bstride)
{
    __shared__ float red[256];
    int row = blockIdx.x;
    int b = row / C, c = row % C;
    const float* p = X + (long)b * bstride + (long)c * HW;
    float s = 0.f;
    for (int t = threadIdx.x; t < HW; t += 256) { float v = p[t]; s += v * v; }
    red[threadIdx.x] = s; __syncthreads();
    for (int o = 128; o; o >>= 1) {
        if (threadIdx.x < o) red[threadIdx.x] += red[threadIdx.x + o];
        __syncthreads();
    }
    if (!threadIdx.x) out[row] = red[0];
}

// l2-normalize rows (clip norm at 1e-12), output contiguous (B,C,HW)
__global__ __launch_bounds__(256)
void l2norm_kernel(const float* __restrict__ X, float* __restrict__ Y,
                   int C, long bstride)
{
    __shared__ float red[256];
    int row = blockIdx.x;
    int b = row / C, c = row % C;
    const float* p = X + (long)b * bstride + (long)c * HW;
    float s = 0.f;
    for (int t = threadIdx.x; t < HW; t += 256) { float v = p[t]; s += v * v; }
    red[threadIdx.x] = s; __syncthreads();
    for (int o = 128; o; o >>= 1) {
        if (threadIdx.x < o) red[threadIdx.x] += red[threadIdx.x + o];
        __syncthreads();
    }
    float nrm = sqrtf(red[0]);
    nrm = nrm > 1e-12f ? nrm : 1e-12f;
    float inv = 1.0f / nrm;
    float* yp = Y + (long)row * HW;
    for (int t = threadIdx.x; t < HW; t += 256) yp[t] = p[t] * inv;
}

// nearest-neighbor distances + keep M rows with smallest NN dist (stable order)
__global__ __launch_bounds__(64)
void matching_select_kernel(const float* __restrict__ x2, const float* __restrict__ c2,
                            const float* __restrict__ xc, int* __restrict__ idx_out)
{
    __shared__ float minv[64];
    __shared__ int   rank[64];
    int b = blockIdx.x;
    int i = threadIdx.x;   // 0..63
    const float* xcb = xc + (long)b * 64 * 64;
    float xi = x2[b * 64 + i];
    float best = 1e30f; int bestj = 0;
    for (int j = 0; j < 64; ++j) {
        float d2 = xi + c2[b * 64 + j] - 2.0f * xcb[i * 64 + j];
        d2 = d2 > 0.f ? d2 : 0.f;
        float dd = sqrtf(d2);
        if (dd < best) { best = dd; bestj = j; }
    }
    minv[i] = best;
    __syncthreads();
    int r = 0;
    for (int j = 0; j < 64; ++j)
        if (minv[j] < best || (minv[j] == best && j < i)) ++r;
    rank[i] = r;
    __syncthreads();
    if (r < MSEL) {
        int slot = 0;
        for (int j = 0; j < i; ++j) if (rank[j] < MSEL) ++slot;
        idx_out[b * MSEL + slot] = bestj;
    }
}

__global__ __launch_bounds__(256)
void gather_rows_kernel(const float* __restrict__ cand, const int* __restrict__ idx,
                        float* __restrict__ out, int row_off)
{
    int bm = blockIdx.y; int b = bm >> 5; int m = bm & 31;
    int t = blockIdx.x * blockDim.x + threadIdx.x;   // < HW
    int src = idx[b * MSEL + m];
    out[((long)b * 64 + row_off + m) * HW + t] = cand[((long)b * 64 + src) * HW + t];
}

// softmax over 32 elems per row, scaled by temperature[head]
__global__ __launch_bounds__(32)
void softmax32_kernel(float* __restrict__ attn, const float* __restrict__ temp)
{
    int row = blockIdx.x;                 // (b*NHEADS + h)*32 + c
    int h = (row >> 5) & (NHEADS - 1);
    float t = temp[h];
    float* p = attn + (long)row * 32;
    int lane = threadIdx.x;
    float v = p[lane] * t;
    float mx = v;
    for (int o = 1; o < 32; o <<= 1) mx = fmaxf(mx, __shfl_xor(mx, o, 32));
    float e = expf(v - mx);
    float s = e;
    for (int o = 1; o < 32; o <<= 1) s += __shfl_xor(s, o, 32);
    p[lane] = e / s;
}

__global__ __launch_bounds__(256)
void add_kernel(const float* __restrict__ a, const float* __restrict__ b,
                float* __restrict__ c, long n)
{
    long i = (long)blockIdx.x * blockDim.x + threadIdx.x;
    if (i < n) c[i] = a[i] + b[i];
}

// ======================= host orchestration =======================
static inline void gemm(hipStream_t s, const float* A, const float* B, const float* Bmul,
                        float* C, const float* bias, int M, int N, int K,
                        int batch, long sA, long sB, long sC, int bT = 0)
{
    dim3 g(N / 16, M / 16, batch);
    if (bT) {
        if (Bmul) gemm_wmma_kernel<1, 1><<<g, 32, 0, s>>>(A, B, Bmul, C, bias, N, K, sA, sB, sC);
        else      gemm_wmma_kernel<1, 0><<<g, 32, 0, s>>>(A, B, Bmul, C, bias, N, K, sA, sB, sC);
    } else {
        if (Bmul) gemm_wmma_kernel<0, 1><<<g, 32, 0, s>>>(A, B, Bmul, C, bias, N, K, sA, sB, sC);
        else      gemm_wmma_kernel<0, 0><<<g, 32, 0, s>>>(A, B, Bmul, C, bias, N, K, sA, sB, sC);
    }
}

extern "C" void kernel_launch(void* const* d_in, const int* in_sizes, int n_in,
                              void* d_out, int out_size, void* d_ws, size_t ws_size,
                              hipStream_t stream)
{
    (void)in_sizes; (void)n_in; (void)out_size; (void)ws_size;
    auto in = [&](int i) { return (const float*)d_in[i]; };
    const float* x          = in(0);
    const float* perception = in(1);
    const float *n1w = in(2), *n1b = in(3), *n2w = in(4), *n2b = in(5);
    const float* temperature = in(6);
    const float *qkv_w = in(7), *qkv_b = in(8), *qkvdw_w = in(9), *qkvdw_b = in(10);
    struct MT { const float *pw,*pb,*d1w,*d1b,*d2w,*d2b,*d3w,*d3b,*cw,*cb; };
    MT amt{in(11),in(12),in(13),in(14),in(15),in(16),in(17),in(18),in(19),in(20)};
    const float *proj_w = in(21), *proj_b = in(22);
    const float *pin1_w = in(23), *pin1_b = in(24), *pin2_w = in(25), *pin2_b = in(26);
    MT fmt{in(27),in(28),in(29),in(30),in(31),in(32),in(33),in(34),in(35),in(36)};
    const float *pout1_w = in(37), *pout1_b = in(38), *pout2_w = in(39), *pout2_b = in(40);

    float* wsp = (float*)d_ws;
    long off = 0;
    auto alloc = [&](long n) { float* p = wsp + off; off += n; return p; };
    const long N1 = (long)Bn * Dc * HW;
    float* p1a = alloc(N1);  float* p2a = alloc(N1);
    float* p1f = alloc(N1);  float* p2f = alloc(N1);
    float* xn   = alloc(N1);
    float* qkv1 = alloc(3 * N1);
    float* qkv2 = alloc(3 * N1);
    float* cat  = alloc(N1);
    float* h1   = alloc(2 * N1);
    float* h2   = alloc(2 * N1);
    float* d3b  = alloc(N1);
    float* mtq  = alloc(N1);
    float* qn   = alloc(N1);
    float* kn   = alloc(N1);
    float* aout = alloc(N1);
    float* tmp1 = alloc(N1);
    float* x1   = alloc(N1);
    float* x2v  = alloc(Bn * Dc);
    float* c2a1 = alloc(Bn * Dc); float* c2a2 = alloc(Bn * Dc);
    float* c2f1 = alloc(Bn * Dc); float* c2f2 = alloc(Bn * Dc);
    float* xc   = alloc(Bn * Dc * Dc);
    float* amat = alloc(Bn * NHEADS * 32 * 32);
    int*   idx  = (int*)alloc(Bn * MSEL);

    // 1) fused perception conv + 8x8 max/avg pool for BOTH matching blocks
    perc_fused_kernel<<<Bn * 64 * 64, 128, 0, stream>>>(perception, amt.pw, amt.pb,
                                                        fmt.pw, fmt.pb, p1a, p2a, p1f, p2f);
    rownorm2_kernel<<<Bn * Dc, 256, 0, stream>>>(p1a, c2a1, Dc, (long)Dc * HW);
    rownorm2_kernel<<<Bn * Dc, 256, 0, stream>>>(p2a, c2a2, Dc, (long)Dc * HW);
    rownorm2_kernel<<<Bn * Dc, 256, 0, stream>>>(p1f, c2f1, Dc, (long)Dc * HW);
    rownorm2_kernel<<<Bn * Dc, 256, 0, stream>>>(p2f, c2f2, Dc, (long)Dc * HW);

    // 2) attention branch
    layernorm_kernel<<<(Bn * HW + 255) / 256, 256, 0, stream>>>(x, n1w, n1b, xn);
    gemm(stream, qkv_w, xn, nullptr, qkv1, qkv_b, 3 * Dc, HW, Dc, Bn,
         0, (long)Dc * HW, (long)3 * Dc * HW);
    gconv3_kernel<<<((long)Bn * 3 * Dc * HW + 255) / 256, 256, 0, stream>>>(
        qkv1, qkvdw_w, qkvdw_b, qkv2, 3 * Dc, 3 * Dc, 3 * Dc, 0);
    const float* qptr = qkv2;
    const float* kptr = qkv2 + (long)Dc * HW;
    const float* vptr = qkv2 + (long)2 * Dc * HW;

    // matching_transformation(q)
    rownorm2_kernel<<<Bn * Dc, 256, 0, stream>>>(qptr, x2v, Dc, (long)3 * Dc * HW);
    gemm(stream, qptr, p1a, nullptr, xc, nullptr, Dc, Dc, HW, Bn,
         (long)3 * Dc * HW, (long)Dc * HW, (long)Dc * Dc, 1);
    matching_select_kernel<<<Bn, 64, 0, stream>>>(x2v, c2a1, xc, idx);
    gather_rows_kernel<<<dim3(HW / 256, Bn * MSEL), 256, 0, stream>>>(p1a, idx, cat, 0);
    gemm(stream, qptr, p2a, nullptr, xc, nullptr, Dc, Dc, HW, Bn,
         (long)3 * Dc * HW, (long)Dc * HW, (long)Dc * Dc, 1);
    matching_select_kernel<<<Bn, 64, 0, stream>>>(x2v, c2a2, xc, idx);
    gather_rows_kernel<<<dim3(HW / 256, Bn * MSEL), 256, 0, stream>>>(p2a, idx, cat, MSEL);
    gemm(stream, amt.d1w, cat, nullptr, h1, amt.d1b, HID, HW, Dc, Bn,
         0, (long)Dc * HW, (long)HID * HW);
    gconv3_kernel<<<((long)Bn * HID * HW + 255) / 256, 256, 0, stream>>>(
        h1, amt.d2w, amt.d2b, h2, HID, HID, HID, 1);
    gemm(stream, amt.d3w, h2, nullptr, d3b, amt.d3b, Dc, HW, HID, Bn,
         0, (long)HID * HW, (long)Dc * HW);
    gemm(stream, amt.cw, d3b, cat, mtq, amt.cb, Dc, HW, Dc, Bn,   // fused d*cat
         0, (long)Dc * HW, (long)Dc * HW);

    // channel attention
    l2norm_kernel<<<Bn * Dc, 256, 0, stream>>>(mtq, qn, Dc, (long)Dc * HW);
    l2norm_kernel<<<Bn * Dc, 256, 0, stream>>>(kptr, kn, Dc, (long)3 * Dc * HW);
    gemm(stream, qn, kn, nullptr, amat, nullptr, 32, 32, HW, Bn * NHEADS,
         (long)32 * HW, (long)32 * HW, 1024, 1);
    softmax32_kernel<<<Bn * NHEADS * 32, 32, 0, stream>>>(amat, temperature);
    for (int b = 0; b < Bn; ++b) {
        gemm(stream, amat + (long)b * NHEADS * 1024, vptr + (long)b * 3 * Dc * HW, nullptr,
             aout + (long)b * Dc * HW, nullptr, 32, HW, 32, NHEADS,
             1024, (long)32 * HW, (long)32 * HW);
    }
    gemm(stream, proj_w, aout, nullptr, tmp1, proj_b, Dc, HW, Dc, Bn,
         0, (long)Dc * HW, (long)Dc * HW);
    add_kernel<<<(N1 + 255) / 256, 256, 0, stream>>>(x, tmp1, x1, N1);

    // 3) feed-forward branch
    layernorm_kernel<<<(Bn * HW + 255) / 256, 256, 0, stream>>>(x1, n2w, n2b, xn);
    gemm(stream, pin1_w, xn, nullptr, h1, pin1_b, HID, HW, Dc, Bn,
         0, (long)Dc * HW, (long)HID * HW);
    gconv3_kernel<<<((long)Bn * Dc * HW + 255) / 256, 256, 0, stream>>>(
        h1, pin2_w, pin2_b, tmp1, HID, Dc, Dc, 0);
    rownorm2_kernel<<<Bn * Dc, 256, 0, stream>>>(tmp1, x2v, Dc, (long)Dc * HW);
    gemm(stream, tmp1, p1f, nullptr, xc, nullptr, Dc, Dc, HW, Bn,
         (long)Dc * HW, (long)Dc * HW, (long)Dc * Dc, 1);
    matching_select_kernel<<<Bn, 64, 0, stream>>>(x2v, c2f1, xc, idx);
    gather_rows_kernel<<<dim3(HW / 256, Bn * MSEL), 256, 0, stream>>>(p1f, idx, cat, 0);
    gemm(stream, tmp1, p2f, nullptr, xc, nullptr, Dc, Dc, HW, Bn,
         (long)Dc * HW, (long)Dc * HW, (long)Dc * Dc, 1);
    matching_select_kernel<<<Bn, 64, 0, stream>>>(x2v, c2f2, xc, idx);
    gather_rows_kernel<<<dim3(HW / 256, Bn * MSEL), 256, 0, stream>>>(p2f, idx, cat, MSEL);
    gemm(stream, fmt.d1w, cat, nullptr, h1, fmt.d1b, HID, HW, Dc, Bn,
         0, (long)Dc * HW, (long)HID * HW);
    gconv3_kernel<<<((long)Bn * HID * HW + 255) / 256, 256, 0, stream>>>(
        h1, fmt.d2w, fmt.d2b, h2, HID, HID, HID, 1);
    gemm(stream, fmt.d3w, h2, nullptr, d3b, fmt.d3b, Dc, HW, HID, Bn,
         0, (long)HID * HW, (long)Dc * HW);
    gemm(stream, fmt.cw, d3b, cat, mtq, fmt.cb, Dc, HW, Dc, Bn,
         0, (long)Dc * HW, (long)Dc * HW);
    gconv3_kernel<<<((long)Bn * HID * HW + 255) / 256, 256, 0, stream>>>(
        mtq, pout1_w, pout1_b, h1, Dc, HID, Dc, 0);
    gemm(stream, pout2_w, h1, nullptr, tmp1, pout2_b, Dc, HW, HID, Bn,
         0, (long)HID * HW, (long)Dc * HW);
    add_kernel<<<(N1 + 255) / 256, 256, 0, stream>>>(x1, tmp1, (float*)d_out, N1);
}
